// OpenProtTransformerBlock_77300821393516
// MI455X (gfx1250) — matrix operand order repaired and checked by reading, not compiled, not run on hardware
//
#include <hip/hip_runtime.h>
#include <cstdint>

typedef __attribute__((ext_vector_type(16))) _Float16 v16h;
typedef __attribute__((ext_vector_type(8)))  float    v8f;

union FragH { v16h v; unsigned u[8]; };

__device__ __forceinline__ unsigned pack_h2(float a, float b) {
  union { _Float16 h[2]; unsigned u; } p;
  p.h[0] = (_Float16)a; p.h[1] = (_Float16)b;
  return p.u;
}

// ---- CDNA5 async global->LDS copy (ASYNCcnt path), with safe fallback ----
#if defined(__HIP_DEVICE_COMPILE__) && __has_builtin(__builtin_amdgcn_global_load_async_to_lds_b128)
#define ASYNC_LDS 1
#else
#define ASYNC_LDS 0
#endif

#if ASYNC_LDS
typedef __attribute__((ext_vector_type(4))) int v4i_;
typedef __attribute__((address_space(1))) v4i_ as1_v4i;
typedef __attribute__((address_space(3))) v4i_ as3_v4i;
#endif

__device__ __forceinline__ void cp_b128(const _Float16* g, _Float16* l) {
#if ASYNC_LDS
  __builtin_amdgcn_global_load_async_to_lds_b128(
      (as1_v4i*)(v4i_*)(uintptr_t)g, (as3_v4i*)(v4i_*)(uintptr_t)l, 0, 0);
#else
  *(uint4*)l = *(const uint4*)g;
#endif
}

__device__ __forceinline__ void async_wait_all() {
#if ASYNC_LDS
#if __has_builtin(__builtin_amdgcn_s_wait_asynccnt)
  __builtin_amdgcn_s_wait_asynccnt(0);
#else
  asm volatile("s_wait_asynccnt 0x0" ::: "memory");
#endif
#endif
}

// ---------------------------------------------------------------------------
// GEMM: C[M,N] = epilogue(A16[M,K] @ BT16[N,K]^T + bias)
// A and B are f16 in global (B pre-transposed to [N][K]).
// Double-buffered LDS fed by async b128 copies; f16 WMMA, fp32 accumulate.
// mode 0: +bias   mode 1: relu(+bias)
// mode 2: res[row,col] + (acc+bias)*gate[row*gateStride + gateOff + col]
// Output: f32 (C32) or f16 (C16) -- exactly one non-null.
// BM=BN=128, BK=32; 256 threads = 8 waves (2x4), wave tile 64x32 (4x2 frags).
// ---------------------------------------------------------------------------
#define BM 128
#define BN 128
#define KP 40   // BK(32) + 8 halves pad; row stride 80B (16B-aligned)

__global__ __launch_bounds__(256)
void gemm_f16wmma(const _Float16* __restrict__ A, const _Float16* __restrict__ BT,
                  const float* __restrict__ bias,
                  float* __restrict__ C32, _Float16* __restrict__ C16,
                  int M, int N, int K, int mode,
                  const float* __restrict__ res,
                  const float* __restrict__ gate, int gateStride, int gateOff)
{
  __shared__ __attribute__((aligned(16))) _Float16 As[2][BM][KP];
  __shared__ __attribute__((aligned(16))) _Float16 Bs[2][BN][KP];
  const int tid  = threadIdx.x;
  const int lane = tid & 31;
  const int wave = tid >> 5;
  const int wm = wave >> 2;      // 0..1
  const int wn = wave & 3;       // 0..3
  const int blockN = blockIdx.x * BN;
  const int blockM = blockIdx.y * BM;
  const _Float16* Abase = A  + (long)blockM * K;
  const _Float16* Bbase = BT + (long)blockN * K;

  v8f acc[4][2];
#pragma unroll
  for (int mf = 0; mf < 4; ++mf)
#pragma unroll
    for (int nf = 0; nf < 2; ++nf)
#pragma unroll
      for (int r = 0; r < 8; ++r) acc[mf][nf][r] = 0.f;

  const int l15 = lane & 15;
  const int kbA = (lane < 16) ? 0 : 8;   // A frag K base per half-wave
  const int kbB = (lane < 16) ? 0 : 8;   // B frag K base per half-wave

  // issue async copies of one k-tile into LDS buffer `buf`
  auto issue_tiles = [&](int buf, int k0) {
#pragma unroll
    for (int i = 0; i < 2; ++i) {
      int idx = tid + 256 * i;           // 0..511
      int row = idx >> 2, c = (idx & 3) * 8;
      cp_b128(Abase + (long)row * K + k0 + c, &As[buf][row][c]);
      cp_b128(Bbase + (long)row * K + k0 + c, &Bs[buf][row][c]);
    }
  };

  const int nsteps = K >> 5;
  issue_tiles(0, 0);
  for (int s = 0; s < nsteps; ++s) {
    const int cur = s & 1;
    async_wait_all();        // own copies into `cur` complete
    __syncthreads();         // all waves' copies done; all reads of `cur^1` done
    if (s + 1 < nsteps) issue_tiles(cur ^ 1, (s + 1) << 5);

    FragH b[2];
#pragma unroll
    for (int nf = 0; nf < 2; ++nf) {
      const _Float16* rp = &Bs[cur][wn * 32 + nf * 16 + l15][0];
#pragma unroll
      for (int d = 0; d < 8; ++d)
        b[nf].u[d] = *(const unsigned*)(rp + kbB + 2 * d + ((d >= 4) ? 8 : 0));
    }
    FragH a[4];
#pragma unroll
    for (int mf = 0; mf < 4; ++mf) {
      const _Float16* rp = &As[cur][wm * 64 + mf * 16 + l15][0];
#pragma unroll
      for (int d = 0; d < 8; ++d)
        a[mf].u[d] = *(const unsigned*)(rp + kbA + 2 * d + ((d >= 4) ? 8 : 0));
    }
#pragma unroll
    for (int mf = 0; mf < 4; ++mf)
#pragma unroll
      for (int nf = 0; nf < 2; ++nf)
        acc[mf][nf] = __builtin_amdgcn_wmma_f32_16x16x32_f16(
            false, a[mf].v, false, b[nf].v, (short)0, acc[mf][nf], false, false);
  }

  const int rOff = (lane < 16) ? 0 : 8;
#pragma unroll
  for (int mf = 0; mf < 4; ++mf) {
#pragma unroll
    for (int nf = 0; nf < 2; ++nf) {
      const int col = blockN + wn * 32 + nf * 16 + l15;
      const float bcol = bias ? bias[col] : 0.f;
#pragma unroll
      for (int r = 0; r < 8; ++r) {
        const int row = blockM + wm * 64 + mf * 16 + r + rOff;
        float v = acc[mf][nf][r] + bcol;
        if (mode == 1) v = v > 0.f ? v : 0.f;
        else if (mode == 2)
          v = res[(long)row * N + col] + v * gate[(long)row * gateStride + gateOff + col];
        if (C16) C16[(long)row * N + col] = (_Float16)v;
        else     C32[(long)row * N + col] = v;
      }
    }
  }
}

// ---------------------------------------------------------------------------
// Flash attention on f16 q/k/v: grid (L/64, B*H), 128 threads = 4 waves.
// S = Q K^T / 8 via WMMA (two K=32 steps), online softmax in fp32 on the
// C-frag layout, P staged through LDS to A-frag layout, O += P V via WMMA.
// Q/K tiles arrive via async b128 copies; V is LDS-transposed on load.
// ---------------------------------------------------------------------------
__global__ __launch_bounds__(128)
void flash_attn(const _Float16* __restrict__ Q, const _Float16* __restrict__ K,
                const _Float16* __restrict__ V, const unsigned char* __restrict__ mask,
                _Float16* __restrict__ O)
{
  const int Lq = 1024, D = 1024;
  __shared__ __attribute__((aligned(16))) _Float16 Qs[64][72];   // [q row][d]
  __shared__ __attribute__((aligned(16))) _Float16 Ks[32][72];   // [key][d]
  __shared__ __attribute__((aligned(16))) _Float16 Vs[64][40];   // [d][key]
  __shared__ __attribute__((aligned(16))) _Float16 Ps[4][16][40];// per-wave P

  const int tid = threadIdx.x, lane = tid & 31, wave = tid >> 5;
  const int b  = blockIdx.y >> 4;
  const int h  = blockIdx.y & 15;
  const int hc = h * 64;
  const int q0 = blockIdx.x * 64;
  const long qtok = (long)b * Lq + q0;

  // Q tile 64x64 (async)
#pragma unroll
  for (int i = 0; i < 4; ++i) {
    int idx = tid + 128 * i;             // 0..511
    int row = idx >> 3, c = (idx & 7) * 8;
    cp_b128(Q + (qtok + row) * D + hc + c, &Qs[row][c]);
  }
  async_wait_all();
  __syncthreads();

  const int l15 = lane & 15;
  const int kbA = (lane < 16) ? 0 : 8;
  const int kbB = (lane < 16) ? 0 : 8;

  FragH aq[2];
#pragma unroll
  for (int s = 0; s < 2; ++s) {
    const _Float16* rp = &Qs[wave * 16 + l15][0];
#pragma unroll
    for (int d = 0; d < 8; ++d)
      aq[s].u[d] = *(const unsigned*)(rp + s * 32 + kbA + 2 * d + ((d >= 4) ? 8 : 0));
  }

  v8f oacc[4];
#pragma unroll
  for (int df = 0; df < 4; ++df)
#pragma unroll
    for (int r = 0; r < 8; ++r) oacc[df][r] = 0.f;
  float mrow[8], lrow[8];
#pragma unroll
  for (int r = 0; r < 8; ++r) { mrow[r] = -3.0e38f; lrow[r] = 0.f; }

  for (int kv0 = 0; kv0 < Lq; kv0 += 32) {
    __syncthreads();   // previous tile fully consumed
    // K tile 32x64 (async)
#pragma unroll
    for (int i = 0; i < 2; ++i) {
      int idx = tid + 128 * i;           // 0..255
      int row = idx >> 3, c = (idx & 7) * 8;
      cp_b128(K + ((long)b * Lq + kv0 + row) * D + hc + c, &Ks[row][c]);
    }
    // V tile 32x64 -> transposed [d][key]
#pragma unroll
    for (int i = 0; i < 2; ++i) {
      int idx = tid + 128 * i;           // 0..255
      int key = idx >> 3, c8 = (idx & 7) * 8;
      union { uint4 u; _Float16 h[8]; } vv;
      vv.u = *(const uint4*)(V + ((long)b * Lq + kv0 + key) * D + hc + c8);
#pragma unroll
      for (int j = 0; j < 8; ++j) Vs[c8 + j][key] = vv.h[j];
    }
    async_wait_all();
    __syncthreads();

    float st[2][8];
#pragma unroll
    for (int nf = 0; nf < 2; ++nf) {
      v8f s;
#pragma unroll
      for (int r = 0; r < 8; ++r) s[r] = 0.f;
#pragma unroll
      for (int ds = 0; ds < 2; ++ds) {
        FragH bk;
        const _Float16* rp = &Ks[nf * 16 + l15][0];
#pragma unroll
        for (int d = 0; d < 8; ++d)
          bk.u[d] = *(const unsigned*)(rp + ds * 32 + kbB + 2 * d + ((d >= 4) ? 8 : 0));
        s = __builtin_amdgcn_wmma_f32_16x16x32_f16(false, aq[ds].v, false, bk.v,
                                                   (short)0, s, false, false);
      }
      const int key = kv0 + nf * 16 + l15;
      const bool ok = mask[(long)b * Lq + key] != 0;
#pragma unroll
      for (int r = 0; r < 8; ++r) st[nf][r] = ok ? s[r] * 0.125f : -3.0e38f;
    }
    // online softmax; each row lives across a 16-lane half (C-frag layout)
#pragma unroll
    for (int r = 0; r < 8; ++r) {
      float tmax = fmaxf(st[0][r], st[1][r]);
#pragma unroll
      for (int m = 1; m < 16; m <<= 1) tmax = fmaxf(tmax, __shfl_xor(tmax, m, 32));
      const float newm = fmaxf(mrow[r], tmax);
      const float corr = __expf(mrow[r] - newm);
      float ps = 0.f;
#pragma unroll
      for (int nf = 0; nf < 2; ++nf) {
        float p = __expf(st[nf][r] - newm);
        st[nf][r] = p; ps += p;
      }
#pragma unroll
      for (int m = 1; m < 16; m <<= 1) ps += __shfl_xor(ps, m, 32);
      lrow[r] = lrow[r] * corr + ps;
      mrow[r] = newm;
#pragma unroll
      for (int df = 0; df < 4; ++df) oacc[df][r] *= corr;
    }
    // stage P (C layout -> LDS -> A-frag layout)
    const int mOff = (lane < 16) ? 0 : 8;
#pragma unroll
    for (int nf = 0; nf < 2; ++nf)
#pragma unroll
      for (int r = 0; r < 8; ++r)
        Ps[wave][r + mOff][nf * 16 + l15] = (_Float16)st[nf][r];
    asm volatile("s_wait_dscnt 0" ::: "memory");
    FragH ap;
    {
      const _Float16* rp = &Ps[wave][l15][0];
#pragma unroll
      for (int d = 0; d < 8; ++d)
        ap.u[d] = *(const unsigned*)(rp + kbA + 2 * d + ((d >= 4) ? 8 : 0));
    }
#pragma unroll
    for (int df = 0; df < 4; ++df) {
      FragH bv;
      const _Float16* rp = &Vs[df * 16 + l15][0];
#pragma unroll
      for (int d = 0; d < 8; ++d)
        bv.u[d] = *(const unsigned*)(rp + kbB + 2 * d + ((d >= 4) ? 8 : 0));
      oacc[df] = __builtin_amdgcn_wmma_f32_16x16x32_f16(false, ap.v, false, bv.v,
                                                        (short)0, oacc[df], false, false);
    }
  }

  const int rOff = (lane < 16) ? 0 : 8;
#pragma unroll
  for (int df = 0; df < 4; ++df) {
    const int col = hc + df * 16 + l15;
#pragma unroll
    for (int r = 0; r < 8; ++r) {
      const long row = qtok + wave * 16 + r + rOff;
      O[row * D + col] = (_Float16)(oacc[df][r] / lrow[r]);
    }
  }
}

// ---------------------------------------------------------------------------
// LayerNorm + adaLN modulation (optional second affine LN); f16 output.
// One block (256 threads) per token row of D=1024.
// ---------------------------------------------------------------------------
__device__ __forceinline__ float block_sum256(float v, float* red) {
  const int tid = threadIdx.x;
  red[tid] = v; __syncthreads();
#pragma unroll
  for (int off = 128; off > 0; off >>= 1) {
    if (tid < off) red[tid] += red[tid + off];
    __syncthreads();
  }
  float r = red[0]; __syncthreads();
  return r;
}

__global__ __launch_bounds__(256)
void ln_mod(const float* __restrict__ x, const float* __restrict__ c,
            _Float16* __restrict__ out, int secondLN,
            const float* __restrict__ g, const float* __restrict__ bln,
            int shiftOff, int scaleOff)
{
  __shared__ float red[256];
  const int D = 1024, SIX = 6144;
  const long row = blockIdx.x;
  const int col = threadIdx.x * 4;
  const float4 xv = *(const float4*)(x + row * D + col);
  float s = xv.x + xv.y + xv.z + xv.w;
  float q = xv.x * xv.x + xv.y * xv.y + xv.z * xv.z + xv.w * xv.w;
  const float mean = block_sum256(s, red) * (1.f / D);
  const float msq  = block_sum256(q, red) * (1.f / D);
  const float rstd = rsqrtf(msq - mean * mean + 1e-5f);
  const float4 sc = *(const float4*)(c + row * SIX + scaleOff + col);
  const float4 sh = *(const float4*)(c + row * SIX + shiftOff + col);
  float h0 = (xv.x - mean) * rstd * (1.f + sc.x) + sh.x;
  float h1 = (xv.y - mean) * rstd * (1.f + sc.y) + sh.y;
  float h2 = (xv.z - mean) * rstd * (1.f + sc.z) + sh.z;
  float h3 = (xv.w - mean) * rstd * (1.f + sc.w) + sh.w;
  if (secondLN) {
    float s2 = h0 + h1 + h2 + h3;
    float q2 = h0 * h0 + h1 * h1 + h2 * h2 + h3 * h3;
    const float m2  = block_sum256(s2, red) * (1.f / D);
    const float mq2 = block_sum256(q2, red) * (1.f / D);
    const float r2  = rsqrtf(mq2 - m2 * m2 + 1e-5f);
    h0 = (h0 - m2) * r2 * g[col + 0] + bln[col + 0];
    h1 = (h1 - m2) * r2 * g[col + 1] + bln[col + 1];
    h2 = (h2 - m2) * r2 * g[col + 2] + bln[col + 2];
    h3 = (h3 - m2) * r2 * g[col + 3] + bln[col + 3];
  }
  unsigned* op = (unsigned*)(out + row * D + col);
  op[0] = pack_h2(h0, h1);
  op[1] = pack_h2(h2, h3);
}

__global__ void silu_f16(const float* __restrict__ x, _Float16* __restrict__ y, long n4) {
  long i = (long)blockIdx.x * blockDim.x + threadIdx.x;
  const long stride = (long)gridDim.x * blockDim.x;
  for (; i < n4; i += stride) {
    float4 v = ((const float4*)x)[i];
    unsigned* op = (unsigned*)(y + i * 4);
    op[0] = pack_h2(v.x / (1.f + __expf(-v.x)), v.y / (1.f + __expf(-v.y)));
    op[1] = pack_h2(v.z / (1.f + __expf(-v.z)), v.w / (1.f + __expf(-v.w)));
  }
}

// W [K][N] fp32 -> WT [N][K] f16, 32x32 LDS tile transpose
__global__ __launch_bounds__(256)
void transpose_cvt(const float* __restrict__ W, _Float16* __restrict__ WT,
                   int K, int N)
{
  __shared__ float t[32][33];
  const int n0 = blockIdx.x * 32, k0 = blockIdx.y * 32;
  const int x = threadIdx.x & 31, y = threadIdx.x >> 5;
#pragma unroll
  for (int j = 0; j < 4; ++j)
    t[y + 8 * j][x] = W[(long)(k0 + y + 8 * j) * N + n0 + x];
  __syncthreads();
#pragma unroll
  for (int j = 0; j < 4; ++j)
    WT[(long)(n0 + y + 8 * j) * K + k0 + x] = (_Float16)t[x][y + 8 * j];
}

// ---------------------------------------------------------------------------
extern "C" void kernel_launch(void* const* d_in, const int* in_sizes, int n_in,
                              void* d_out, int out_size, void* d_ws, size_t ws_size,
                              hipStream_t stream)
{
  (void)n_in; (void)out_size; (void)ws_size;
  const int B = 2, L = 1024, D = 1024, SIX = 6144, FF = 4096;
  const int BT = B * L;

  const float* x      = (const float*)d_in[0];
  const float* z      = (const float*)d_in[1];
  const float* trans  = (const float*)d_in[2];
  const unsigned char* mask = (const unsigned char*)d_in[3];
  const float* x_cond = (const float*)d_in[4];
  const float* ada_W  = (const float*)d_in[5];
  const float* ada_b  = (const float*)d_in[6];
  const float* Wq = (const float*)d_in[7];   const float* bq = (const float*)d_in[8];
  const float* Wk = (const float*)d_in[9];   const float* bk = (const float*)d_in[10];
  const float* Wv = (const float*)d_in[11];  const float* bv = (const float*)d_in[12];
  const float* Wo = (const float*)d_in[13];  const float* bo = (const float*)d_in[14];
  const float* ln_g = (const float*)d_in[15];
  const float* ln_b = (const float*)d_in[16];
  const float* W1 = (const float*)d_in[17];  const float* b1 = (const float*)d_in[18];
  const float* W2 = (const float*)d_in[19];  const float* b2 = (const float*)d_in[20];

  float* xout = (float*)d_out;
  float* zout = xout + (long)BT * D;
  float* tout = zout + (long)in_sizes[1];

  // z and trans pass through untouched
  (void)hipMemcpyAsync(zout, z, (size_t)in_sizes[1] * sizeof(float),
                       hipMemcpyDeviceToDevice, stream);
  (void)hipMemcpyAsync(tout, trans, (size_t)in_sizes[2] * sizeof(float),
                       hipMemcpyDeviceToDevice, stream);

  // fp32 workspace
  float* wsf = (float*)d_ws;
  float* c_  = wsf;  wsf += (long)BT * SIX;   // adaLN modulation (fp32)
  float* x1_ = wsf;  wsf += (long)BT * D;     // post-attention residual (fp32)
  // f16 workspace
  _Float16* wh   = (_Float16*)wsf;
  _Float16* adaT = wh;  wh += (long)SIX * D;  // ada_W^T
  _Float16* wqT  = wh;  wh += (long)D * D;
  _Float16* wkT  = wh;  wh += (long)D * D;
  _Float16* wvT  = wh;  wh += (long)D * D;
  _Float16* woT  = wh;  wh += (long)D * D;
  _Float16* w1T  = wh;  wh += (long)FF * D;   // W1^T [FF][D]
  _Float16* w2T  = wh;  wh += (long)D * FF;   // W2^T [D][FF]
  _Float16* s16  = wh;  wh += (long)BT * D;   // silu(x_cond)
  _Float16* h16  = wh;  wh += (long)BT * D;   // normed/modulated activations
  _Float16* q16  = wh;  wh += (long)BT * D;
  _Float16* k16  = wh;  wh += (long)BT * D;
  _Float16* v16  = wh;  wh += (long)BT * D;
  _Float16* o16  = wh;  wh += (long)BT * D;   // attention out (pre-Wo)
  _Float16* f16b = wh;  wh += (long)BT * FF;  // FFN hidden

  // weight transpose+convert (once per launch)
  transpose_cvt<<<dim3(SIX / 32, D / 32), 256, 0, stream>>>(ada_W, adaT, D, SIX);
  transpose_cvt<<<dim3(D / 32, D / 32), 256, 0, stream>>>(Wq, wqT, D, D);
  transpose_cvt<<<dim3(D / 32, D / 32), 256, 0, stream>>>(Wk, wkT, D, D);
  transpose_cvt<<<dim3(D / 32, D / 32), 256, 0, stream>>>(Wv, wvT, D, D);
  transpose_cvt<<<dim3(D / 32, D / 32), 256, 0, stream>>>(Wo, woT, D, D);
  transpose_cvt<<<dim3(FF / 32, D / 32), 256, 0, stream>>>(W1, w1T, D, FF);
  transpose_cvt<<<dim3(D / 32, FF / 32), 256, 0, stream>>>(W2, w2T, FF, D);

  silu_f16<<<1024, 256, 0, stream>>>(x_cond, s16, (long)BT * D / 4);

  // c = silu(x_cond) @ ada_W + ada_b   [2048 x 6144] fp32
  gemm_f16wmma<<<dim3(SIX / 128, BT / 128), 256, 0, stream>>>(
      s16, adaT, ada_b, c_, nullptr, BT, SIX, D, 0, nullptr, nullptr, 0, 0);

  // h = ln(x) * (1 + scale_mha) + shift_mha  (f16)
  ln_mod<<<BT, 256, 0, stream>>>(x, c_, h16, 0, nullptr, nullptr, 0, D);

  gemm_f16wmma<<<dim3(D / 128, BT / 128), 256, 0, stream>>>(
      h16, wqT, bq, nullptr, q16, BT, D, D, 0, nullptr, nullptr, 0, 0);
  gemm_f16wmma<<<dim3(D / 128, BT / 128), 256, 0, stream>>>(
      h16, wkT, bk, nullptr, k16, BT, D, D, 0, nullptr, nullptr, 0, 0);
  gemm_f16wmma<<<dim3(D / 128, BT / 128), 256, 0, stream>>>(
      h16, wvT, bv, nullptr, v16, BT, D, D, 0, nullptr, nullptr, 0, 0);

  flash_attn<<<dim3(L / 64, B * 16), 128, 0, stream>>>(q16, k16, v16, mask, o16);

  // x1 = x + (o @ Wo + bo) * gate_mha  (fp32)
  gemm_f16wmma<<<dim3(D / 128, BT / 128), 256, 0, stream>>>(
      o16, woT, bo, x1_, nullptr, BT, D, D, 2, x, c_, SIX, 2 * D);

  // h = ln_affine( ln(x1)*(1+scale_mlp)+shift_mlp , g, b )  (f16)
  ln_mod<<<BT, 256, 0, stream>>>(x1_, c_, h16, 1, ln_g, ln_b, 3 * D, 4 * D);

  // f = relu(h @ W1 + b1)  (f16)
  gemm_f16wmma<<<dim3(FF / 128, BT / 128), 256, 0, stream>>>(
      h16, w1T, b1, nullptr, f16b, BT, FF, D, 1, nullptr, nullptr, 0, 0);

  // x_out = x1 + (f @ W2 + b2) * gate_mlp  (fp32)
  gemm_f16wmma<<<dim3(D / 128, BT / 128), 256, 0, stream>>>(
      f16b, w2T, b2, xout, nullptr, BT, D, FF, 2, x1_, c_, SIX, 5 * D);
}
